// GridMPNNLayer_81896436400372
// MI455X (gfx1250) — matrix operand order repaired and checked by reading, not compiled, not run on hardware
//
#include <hip/hip_runtime.h>

#define GRID_N 10
#define NN 100
#define H 128
#define E_EDGES 360
#define E_PAD 368
#define ETILES 23
#define MTILES 7
#define BATCH 2048
#define LN_EPS 1e-5f

typedef __attribute__((ext_vector_type(8)))  __bf16 v8bf;
typedef __attribute__((ext_vector_type(16))) __bf16 v16bf;
typedef __attribute__((ext_vector_type(8)))  float  v8f;

// element strides
#define XS  136   // bf16 activation rows (x, m-scratch, u1)
#define WS  136   // transposed bf16 weight rows (K=128)
#define WUS 264   // transposed bf16 weight rows (K=256)
#define AS  132   // f32 agg / h rows

// byte offsets into dynamic LDS
#define OFF_WA    0        // phase AB: w1t ; phase CD: wu1t
#define OFF_W2    34816    // phase AB: w2t
#define OFF_WU2   67584    // phase CD: wu2t
#define OFF_XBF   102400   // 112 x 136 bf16
#define OFF_AGG   132864   // 112 x 132 f32  (reused as h)
#define OFF_MS    192000   // union: 8 waves x 16x136 bf16 m-scratch / 112x136 bf16 u1
#define OFF_EDROW 226816
#define OFF_B1    227328
#define OFF_B2    227840
#define OFF_BU1   228352
#define OFF_BU2   228864
#define OFF_GAMMA 229376
#define OFF_BETA  229888
#define OFF_SRC   230400
#define OFF_DST   231872
#define OFF_EDV   233344
#define SMEM_BYTES 234816

__device__ __forceinline__ v16bf cat8(v8bf c0, v8bf c1) {
  v16bf a;
#pragma unroll
  for (int i = 0; i < 8; i++) { a[i] = c0[i]; a[i + 8] = c1[i]; }
  return a;
}

// A fragment (16x32 tile): lane = m(0..15) + 16*kg ; per-lane K chunks
// [kbase + kg*8 .. +7] and [kbase + 16 + kg*8 .. +7]
__device__ __forceinline__ v16bf loadA_bf(const __bf16* row, int kbase, int kg) {
  v8bf c0 = *(const v8bf*)(row + kbase + kg * 8);
  v8bf c1 = *(const v8bf*)(row + kbase + 16 + kg * 8);
  return cat8(c0, c1);
}

__device__ __forceinline__ v16bf loadA_f32(const float* row, int kbase, int kg) {
  const float* p0 = row + kbase + kg * 8;
  const float* p1 = row + kbase + 16 + kg * 8;
  v16bf a;
#pragma unroll
  for (int i = 0; i < 8; i++) { a[i] = (__bf16)p0[i]; a[i + 8] = (__bf16)p1[i]; }
  return a;
}

// B fragment (32x16 tile) from transposed weight wt[n*stride + k]:
// lane = n(0..15) + 16*kh ; per-lane K chunks [kbase + kh*8 .. +7] and [kbase + 16 + kh*8 .. +7]
__device__ __forceinline__ v16bf loadB_bf(const __bf16* wt, int stride, int n,
                                          int kbase, int kh) {
  const __bf16* p = wt + n * stride + kbase + kh * 8;
  v8bf c0 = *(const v8bf*)(p);
  v8bf c1 = *(const v8bf*)(p + 16);
  return cat8(c0, c1);
}

__device__ __forceinline__ v8f wmma_bf16(v16bf a, v16bf b, v8f c) {
  return __builtin_amdgcn_wmma_f32_16x16x32_bf16(false, a, false, b, (short)0, c,
                                                 false, false);
}

__global__ __launch_bounds__(256) void mpnn_fused_kernel(
    const float* __restrict__ xg_all, const int* __restrict__ srcI,
    const int* __restrict__ dstI, const float* __restrict__ edirI,
    const float* __restrict__ w_msg1, const float* __restrict__ b_msg1,
    const float* __restrict__ w_msg2, const float* __restrict__ b_msg2,
    const float* __restrict__ w_upd1, const float* __restrict__ b_upd1,
    const float* __restrict__ w_upd2, const float* __restrict__ b_upd2,
    const float* __restrict__ gammaI, const float* __restrict__ betaI,
    float* __restrict__ outg) {
  extern __shared__ char smem[];
  const int tid = threadIdx.x;
  const int b = blockIdx.x;
  const float* xg = xg_all + (size_t)b * NN * H;

  __bf16* xbf  = (__bf16*)(smem + OFF_XBF);
  float*  aggf = (float*)(smem + OFF_AGG);
  float*  edrow = (float*)(smem + OFF_EDROW);
  float*  b1f = (float*)(smem + OFF_B1);
  float*  b2f = (float*)(smem + OFF_B2);
  float*  bu1f = (float*)(smem + OFF_BU1);
  float*  bu2f = (float*)(smem + OFF_BU2);
  float*  gmf = (float*)(smem + OFF_GAMMA);
  float*  btf = (float*)(smem + OFF_BETA);
  int*    SRC = (int*)(smem + OFF_SRC);
  int*    DST = (int*)(smem + OFF_DST);
  float*  EDV = (float*)(smem + OFF_EDV);

  // ---- Phase 0: stage x, zero agg, stage msg weights (transposed bf16) ----
  {
    __bf16* w1t = (__bf16*)(smem + OFF_WA);
    __bf16* w2t = (__bf16*)(smem + OFF_W2);
    for (int i = tid; i < 112 * 32; i += 256) {
      int row = i >> 5, c4 = (i & 31) * 4;
      float4 v = make_float4(0.f, 0.f, 0.f, 0.f);
      if (row < NN) v = *(const float4*)(xg + row * H + c4);
      __bf16* d = xbf + row * XS + c4;
      d[0] = (__bf16)v.x; d[1] = (__bf16)v.y; d[2] = (__bf16)v.z; d[3] = (__bf16)v.w;
    }
    for (int i = tid; i < 112 * AS; i += 256) aggf[i] = 0.f;
    for (int i = tid; i < H * H; i += 256) {
      int k = i >> 7, n = i & 127;
      w1t[n * WS + k] = (__bf16)w_msg1[k * H + n];
      w2t[n * WS + k] = (__bf16)w_msg2[k * H + n];
    }
    for (int i = tid; i < H; i += 256) {
      edrow[i] = w_msg1[H * H + i];  // row 128 of (129,128): edge_dir feature
      b1f[i] = b_msg1[i]; b2f[i] = b_msg2[i];
      bu1f[i] = b_upd1[i]; bu2f[i] = b_upd2[i];
      gmf[i] = gammaI[i]; btf[i] = betaI[i];
    }
    for (int e = tid; e < E_PAD; e += 256) {
      if (e < E_EDGES) { SRC[e] = srcI[e]; DST[e] = dstI[e]; EDV[e] = edirI[e]; }
      else { SRC[e] = 0; DST[e] = 0; EDV[e] = 0.f; }
    }
  }
  __syncthreads();

  const int w = tid >> 5, lane = tid & 31;
  const int ml = lane & 15, kg = lane >> 4;

  // ---- Phase AB: messages (msg1 -> relu -> msg2) + scatter-add into agg ----
  {
    const __bf16* w1t = (const __bf16*)(smem + OFF_WA);
    const __bf16* w2t = (const __bf16*)(smem + OFF_W2);
    __bf16* ms = (__bf16*)(smem + OFF_MS) + w * (16 * XS);

    for (int et = w; et < ETILES; et += 8) {
      int ea = et * 16 + ml;
      int srow = SRC[ea];  // padded edges map to row 0 (result discarded)
      const __bf16* xrow = xbf + srow * XS;
      v16bf a1[4];
#pragma unroll
      for (int kb = 0; kb < 4; kb++) a1[kb] = loadA_bf(xrow, kb * 32, kg);

      int erow0 = et * 16 + 8 * kg;  // C rows this lane owns
      float edv[8]; int dstr[8]; bool ev[8];
#pragma unroll
      for (int r = 0; r < 8; r++) {
        int e = erow0 + r;
        edv[r] = EDV[e]; dstr[r] = DST[e]; ev[r] = (e < E_EDGES);
      }

      // msg1: m = relu(x_src @ W1[0:128] + ed * W1[128] + b1)
#pragma unroll
      for (int nt = 0; nt < 8; nt++) {
        v8f c;
#pragma unroll
        for (int i = 0; i < 8; i++) c[i] = 0.f;
#pragma unroll
        for (int kb = 0; kb < 4; kb++)
          c = wmma_bf16(a1[kb], loadB_bf(w1t, WS, nt * 16 + ml, kb * 32, kg), c);
        int n = nt * 16 + ml;
        float bn = b1f[n], ern = edrow[n];
#pragma unroll
        for (int r = 0; r < 8; r++) {
          float v = c[r] + bn + edv[r] * ern;
          ms[(r + 8 * kg) * XS + n] = (__bf16)(v > 0.f ? v : 0.f);
        }
      }

      // msg2: messages = m @ W2 + b2, scatter-add to agg[dst]
      v16bf a2[4];
#pragma unroll
      for (int kb = 0; kb < 4; kb++) a2[kb] = loadA_bf(ms + ml * XS, kb * 32, kg);
#pragma unroll
      for (int nt = 0; nt < 8; nt++) {
        v8f c;
#pragma unroll
        for (int i = 0; i < 8; i++) c[i] = 0.f;
#pragma unroll
        for (int kb = 0; kb < 4; kb++)
          c = wmma_bf16(a2[kb], loadB_bf(w2t, WS, nt * 16 + ml, kb * 32, kg), c);
        int n = nt * 16 + ml;
        float bn = b2f[n];
#pragma unroll
        for (int r = 0; r < 8; r++)
          if (ev[r]) atomicAdd(&aggf[dstr[r] * AS + n], c[r] + bn);
      }
    }
  }
  __syncthreads();

  // ---- restage update weights over the msg-weight region ----
  {
    __bf16* wu1t = (__bf16*)(smem + OFF_WA);
    __bf16* wu2t = (__bf16*)(smem + OFF_WU2);
    for (int i = tid; i < 256 * H; i += 256) {
      int k = i >> 7, n = i & 127;
      wu1t[n * WUS + k] = (__bf16)w_upd1[k * H + n];
    }
    for (int i = tid; i < H * H; i += 256) {
      int k = i >> 7, n = i & 127;
      wu2t[n * WS + k] = (__bf16)w_upd2[k * H + n];
    }
  }
  __syncthreads();

  // ---- Phase CD: u1 = relu([x,agg]@Wu1 + b), u2 = u1@Wu2 + b, h = x + u2 ----
  float* hf = aggf;  // reuse agg region for h (per-wave rows: read-then-write)
  {
    const __bf16* wu1t = (const __bf16*)(smem + OFF_WA);
    const __bf16* wu2t = (const __bf16*)(smem + OFF_WU2);
    __bf16* u1bf = (__bf16*)(smem + OFF_MS);
    if (w < MTILES) {
      int mt = w;
      int row = mt * 16 + ml;
      v16bf ax[4], ag[4];
#pragma unroll
      for (int kb = 0; kb < 4; kb++) {
        ax[kb] = loadA_bf(xbf + row * XS, kb * 32, kg);
        ag[kb] = loadA_f32(aggf + row * AS, kb * 32, kg);
      }
#pragma unroll
      for (int nt = 0; nt < 8; nt++) {
        v8f c;
#pragma unroll
        for (int i = 0; i < 8; i++) c[i] = 0.f;
#pragma unroll
        for (int kb = 0; kb < 4; kb++)
          c = wmma_bf16(ax[kb], loadB_bf(wu1t, WUS, nt * 16 + ml, kb * 32, kg), c);
#pragma unroll
        for (int kb = 0; kb < 4; kb++)
          c = wmma_bf16(ag[kb], loadB_bf(wu1t, WUS, nt * 16 + ml, 128 + kb * 32, kg), c);
        int n = nt * 16 + ml;
        float bn = bu1f[n];
#pragma unroll
        for (int r = 0; r < 8; r++) {
          float v = c[r] + bn;
          u1bf[(mt * 16 + r + 8 * kg) * XS + n] = (__bf16)(v > 0.f ? v : 0.f);
        }
      }
      v16bf au[4];
#pragma unroll
      for (int kb = 0; kb < 4; kb++) au[kb] = loadA_bf(u1bf + row * XS, kb * 32, kg);
#pragma unroll
      for (int nt = 0; nt < 8; nt++) {
        v8f c;
#pragma unroll
        for (int i = 0; i < 8; i++) c[i] = 0.f;
#pragma unroll
        for (int kb = 0; kb < 4; kb++)
          c = wmma_bf16(au[kb], loadB_bf(wu2t, WS, nt * 16 + ml, kb * 32, kg), c);
        int n = nt * 16 + ml;
        float bn = bu2f[n];
#pragma unroll
        for (int r = 0; r < 8; r++) {
          int node = mt * 16 + r + 8 * kg;
          if (node < NN) {
            float xr = xg[node * H + n];       // residual from global (L2-hot)
            hf[node * AS + n] = c[r] + bn + xr;
          }
        }
      }
    }
  }
  __syncthreads();

  // ---- Phase E: LayerNorm, one wave per row, lane handles 4 columns ----
  for (int rrow = w; rrow < NN; rrow += 8) {
    const float* hr = hf + rrow * AS;
    float4 v = *(const float4*)(hr + lane * 4);
    float s = v.x + v.y + v.z + v.w;
#pragma unroll
    for (int off = 16; off > 0; off >>= 1) s += __shfl_xor(s, off, 32);
    float mu = s * (1.0f / H);
    float dx = v.x - mu, dy = v.y - mu, dz = v.z - mu, dw = v.w - mu;
    float q = dx * dx + dy * dy + dz * dz + dw * dw;
#pragma unroll
    for (int off = 16; off > 0; off >>= 1) q += __shfl_xor(q, off, 32);
    float rstd = rsqrtf(q * (1.0f / H) + LN_EPS);
    int cb = lane * 4;
    float4 o;
    o.x = dx * rstd * gmf[cb + 0] + btf[cb + 0];
    o.y = dy * rstd * gmf[cb + 1] + btf[cb + 1];
    o.z = dz * rstd * gmf[cb + 2] + btf[cb + 2];
    o.w = dw * rstd * gmf[cb + 3] + btf[cb + 3];
    *(float4*)(outg + ((size_t)b * NN + rrow) * H + cb) = o;
  }
}

extern "C" void kernel_launch(void* const* d_in, const int* in_sizes, int n_in,
                              void* d_out, int out_size, void* d_ws, size_t ws_size,
                              hipStream_t stream) {
  (void)in_sizes; (void)n_in; (void)out_size; (void)d_ws; (void)ws_size;
  const float* x   = (const float*)d_in[0];
  const int*   src = (const int*)d_in[1];
  const int*   dst = (const int*)d_in[2];
  const float* ed  = (const float*)d_in[3];
  const float* w1  = (const float*)d_in[4];
  const float* bm1 = (const float*)d_in[5];
  const float* w2  = (const float*)d_in[6];
  const float* bm2 = (const float*)d_in[7];
  const float* wu1 = (const float*)d_in[8];
  const float* bu1 = (const float*)d_in[9];
  const float* wu2 = (const float*)d_in[10];
  const float* bu2 = (const float*)d_in[11];
  const float* gm  = (const float*)d_in[12];
  const float* bt  = (const float*)d_in[13];
  mpnn_fused_kernel<<<BATCH, 256, SMEM_BYTES, stream>>>(
      x, src, dst, ed, w1, bm1, w2, bm2, wu1, bu1, wu2, bu2, gm, bt,
      (float*)d_out);
}
